// CONV_GNN_36910948942071
// MI455X (gfx1250) — matrix operand order repaired and checked by reading, not compiled, not run on hardware
//
#include <hip/hip_runtime.h>

typedef __attribute__((ext_vector_type(16))) __bf16       v16bf;
typedef __attribute__((ext_vector_type(8)))  float        v8f;
typedef __attribute__((ext_vector_type(4)))  unsigned int u32x4;
typedef __attribute__((ext_vector_type(4)))  float        f32x4;
typedef __attribute__((ext_vector_type(4)))  __bf16       bf16x4;

union ABfrag { u32x4 u[2]; v16bf v; };

#define FH 128   // F_HID

// Hardware no-return f32 atomic add, device scope (visible across WGPs).
// Tracked on STOREcnt; s_endpgm's implicit wait-idle drains it.
__device__ __forceinline__ void gadd_f32(float* p, float v) {
  asm volatile("global_atomic_add_f32 %0, %1, off scope:SCOPE_DEV"
               : : "v"(p), "v"(v) : "memory");
}

// ---------- degree / norm ----------
__global__ void k_deg_init(float* deg, int n) {
  int i = blockIdx.x * blockDim.x + threadIdx.x;
  if (i < n) deg[i] = 1.0f;                    // self-loop contributes 1
}
__global__ void k_deg_acc(const int* __restrict__ dst, float* deg, int e) {
  int i = blockIdx.x * blockDim.x + threadIdx.x;
  if (i < e) gadd_f32(&deg[dst[i]], 1.0f);
}
__global__ void k_deg_fin(float* deg, int n) {
  int i = blockIdx.x * blockDim.x + threadIdx.x;
  if (i < n) deg[i] = rsqrtf(deg[i]);          // deg >= 1 always
}

// ---------- f32 -> bf16 conversion (x), vectorized x4 ----------
__global__ void k_cvt_bf16(const float* __restrict__ in, __bf16* __restrict__ out, int n4) {
  int i = blockIdx.x * blockDim.x + threadIdx.x;
  if (i >= n4) return;
  f32x4 v = ((const f32x4*)in)[i];
  bf16x4 o;
  o.x = (__bf16)v.x; o.y = (__bf16)v.y; o.z = (__bf16)v.z; o.w = (__bf16)v.w;
  ((bf16x4*)out)[i] = o;
}

// ---------- pack weights [K,128] f32 -> WMMA B-lane order bf16 ----------
// packed index t = ((c*nK + s)*32 + lane)*16 + j
//   lane 0-15 : col = 16c+lane,    K = s*32 + j (j<8), s*32 + 16 + (j-8) (j>=8)
//   lane16-31 : col = 16c+lane-16, K = s*32 + 8 + j (j<8), s*32 + 24 + (j-8) (j>=8)
__global__ void k_pack_w(const float* __restrict__ W, __bf16* __restrict__ wpk, int Kdim) {
  int t = blockIdx.x * blockDim.x + threadIdx.x;
  int total = Kdim * FH;
  if (t >= total) return;
  int j    = t & 15;
  int lane = (t >> 4) & 31;
  int q    = t >> 9;
  int nK   = Kdim >> 5;
  int s    = q % nK;
  int c    = q / nK;
  int k    = s * 32 + ((j >> 3) << 4) + ((lane >> 4) << 3) + (j & 7);
  int col  = (c << 4) + (lane & 15);
  wpk[t] = (__bf16)W[k * FH + col];
}

// ---------- WMMA GEMM: C[N,128] = A[N,K](bf16) * Wpk(bf16), f32 accum ----------
__global__ __launch_bounds__(128) void k_gemm(const __bf16* __restrict__ A,
                                              const __bf16* __restrict__ Wpk,
                                              float* __restrict__ C,
                                              int N, int Kdim) {
  int lane    = threadIdx.x & 31;
  int wave    = threadIdx.x >> 5;
  int rowBase = blockIdx.x * 64 + wave * 16;
  int row     = rowBase + (lane & 15);
  int rowc    = (row < N) ? row : (N - 1);   // clamp; OOB strips never stored
  int nK      = Kdim >> 5;

  v8f acc[8] = {};
  // A: row-major; lane half selects K sub-block per ISA 16-bit A layout
  const __bf16* ap = A + (long)rowc * Kdim + ((lane >> 4) << 3);

  for (int s = 0; s < nK; ++s) {
    ABfrag a;
    a.u[0] = *(const u32x4*)(ap + s * 32);        // K = k0 + base .. +7
    a.u[1] = *(const u32x4*)(ap + s * 32 + 16);   // K = k0 + 16 + base .. +7
#pragma unroll
    for (int c = 0; c < 8; ++c) {
      const __bf16* bp = Wpk + ((((long)c * nK + s) * 32 + lane) << 4);
      ABfrag b;
      b.u[0] = *(const u32x4*)bp;
      b.u[1] = *(const u32x4*)(bp + 8);
      acc[c] = __builtin_amdgcn_wmma_f32_16x16x32_bf16(
          false, a.v, false, b.v, (short)0, acc[c], false, false);
    }
  }

  // C/D layout: VGPR v, lanes 0-15 -> M=v, lanes 16-31 -> M=v+8 ; N = lane&15
  int rstore = rowBase + ((lane >> 4) << 3);
#pragma unroll
  for (int c = 0; c < 8; ++c) {
    int col = c * 16 + (lane & 15);
#pragma unroll
    for (int v = 0; v < 8; ++v) {
      int rr = rstore + v;
      if (rr < N) C[(long)rr * FH + col] = acc[c][v];
    }
  }
}

// ---------- self-loop init: out[n,:] = gemm[n,:] * dis[n]^2 ----------
__global__ void k_selfloop(const float* __restrict__ g, const float* __restrict__ dis,
                           float* __restrict__ out, int n32) {
  int t = blockIdx.x * blockDim.x + threadIdx.x;
  if (t >= n32) return;
  int n = t >> 5;
  float w = dis[n]; w *= w;
  f32x4 v = ((const f32x4*)g)[t];
  v.x *= w; v.y *= w; v.z *= w; v.w *= w;
  ((f32x4*)out)[t] = v;
}

// ---------- edge scatter: one wave per edge, float4 per lane, hw f32 atomics ----------
__global__ void k_edges(const float* __restrict__ g,
                        const int* __restrict__ src, const int* __restrict__ dst,
                        const float* __restrict__ dis, float* __restrict__ out, int E) {
  int t = blockIdx.x * blockDim.x + threadIdx.x;
  int e = t >> 5, lane = t & 31;
  if (e >= E) return;
  int s = src[e], d = dst[e];
  float w = dis[s] * dis[d];
  f32x4 v = ((const f32x4*)(g + (long)s * FH))[lane];   // coalesced 512B/wave
  float* o = out + (long)d * FH + lane * 4;
  gadd_f32(o + 0, v.x * w);
  gadd_f32(o + 1, v.y * w);
  gadd_f32(o + 2, v.z * w);
  gadd_f32(o + 3, v.w * w);
}

// ---------- bias + ReLU, emit bf16 activations (or final f32) ----------
__global__ void k_finalize(const float* __restrict__ agg, const float* __restrict__ bias,
                           __bf16* __restrict__ hnext, float* __restrict__ outF,
                           int n32, int last) {
  int t = blockIdx.x * blockDim.x + threadIdx.x;
  if (t >= n32) return;
  int base = t * 4;
  int col  = base & (FH - 1);
  f32x4 v = *(const f32x4*)(agg + base);
  f32x4 b = *(const f32x4*)(bias + col);
  v.x = fmaxf(v.x + b.x, 0.f); v.y = fmaxf(v.y + b.y, 0.f);
  v.z = fmaxf(v.z + b.z, 0.f); v.w = fmaxf(v.w + b.w, 0.f);
  if (last) {
    *(f32x4*)(outF + base) = v;
  } else {
    bf16x4 o;
    o.x = (__bf16)v.x; o.y = (__bf16)v.y; o.z = (__bf16)v.z; o.w = (__bf16)v.w;
    *(bf16x4*)(hnext + base) = o;
  }
}

extern "C" void kernel_launch(void* const* d_in, const int* in_sizes, int n_in,
                              void* d_out, int out_size, void* d_ws, size_t ws_size,
                              hipStream_t stream) {
  const int FIN = 256;
  int N = in_sizes[0] / FIN;          // 50000
  int E = in_sizes[1] / 2;            // 640000

  const float* x   = (const float*)d_in[0];
  const int*   src = (const int*)d_in[1];
  const int*   dst = src + E;
  const float* Ws[5] = {(const float*)d_in[2], (const float*)d_in[4], (const float*)d_in[6],
                        (const float*)d_in[8], (const float*)d_in[10]};
  const float* Bs[5] = {(const float*)d_in[3], (const float*)d_in[5], (const float*)d_in[7],
                        (const float*)d_in[9], (const float*)d_in[11]};
  int Ks[5] = {FIN, FH, FH, FH, FH};

  char* ws = (char*)d_ws;
  auto rnd = [](size_t x) { return (x + 255) & ~(size_t)255; };
  size_t off = 0;
  float*  dis  = (float*)(ws + off);  off += rnd((size_t)N * 4);
  __bf16* hbf  = (__bf16*)(ws + off); off += rnd((size_t)N * FIN * 2);
  float*  gbuf = (float*)(ws + off);  off += rnd((size_t)N * FH * 4);
  float*  abuf = (float*)(ws + off);  off += rnd((size_t)N * FH * 4);
  __bf16* wpk  = (__bf16*)(ws + off);
  size_t wo[5]; wo[0] = 0;
  for (int i = 1; i < 5; ++i) wo[i] = wo[i - 1] + (size_t)Ks[i - 1] * FH;

  const int T = 256;
  // normalization
  k_deg_init<<<(N + T - 1) / T, T, 0, stream>>>(dis, N);
  k_deg_acc <<<(E + T - 1) / T, T, 0, stream>>>(dst, dis, E);
  k_deg_fin <<<(N + T - 1) / T, T, 0, stream>>>(dis, N);
  // activations -> bf16
  k_cvt_bf16<<<((N * FIN / 4) + T - 1) / T, T, 0, stream>>>(x, hbf, N * FIN / 4);
  // pack weights into WMMA B-lane order
  for (int i = 0; i < 5; ++i)
    k_pack_w<<<((Ks[i] * FH) + T - 1) / T, T, 0, stream>>>(Ws[i], wpk + wo[i], Ks[i]);

  int n32 = N * (FH / 4);
  for (int l = 0; l < 5; ++l) {
    k_gemm<<<(N + 63) / 64, 128, 0, stream>>>(hbf, wpk + wo[l], gbuf, N, Ks[l]);
    k_selfloop<<<(n32 + T - 1) / T, T, 0, stream>>>(gbuf, dis, abuf, n32);
    k_edges<<<((E * 32) + T - 1) / T, T, 0, stream>>>(gbuf, src, dst, dis, abuf, E);
    k_finalize<<<(n32 + T - 1) / T, T, 0, stream>>>(abuf, Bs[l], hbf, (float*)d_out,
                                                    n32, l == 4 ? 1 : 0);
  }
}
